// StrucNet_59725815218503
// MI455X (gfx1250) — matrix operand order repaired and checked by reading, not compiled, not run on hardware
//
#include <hip/hip_runtime.h>
#include <stdint.h>

#define N_  12288
#define D_  512
#define H_  1024
#define KTOP 32

typedef __attribute__((ext_vector_type(16))) __bf16 v16bf;
typedef __attribute__((ext_vector_type(8)))  float  v8f;

// ---------------------------------------------------------------------------
// f32 -> bf16 conversion (grid-stride)
// ---------------------------------------------------------------------------
__global__ __launch_bounds__(256)
void k_cvt(const float* __restrict__ s, __bf16* __restrict__ d, size_t n) {
    size_t i = (size_t)blockIdx.x * blockDim.x + threadIdx.x;
    size_t stride = (size_t)gridDim.x * blockDim.x;
    for (; i < n; i += stride) d[i] = (__bf16)s[i];
}

// ---------------------------------------------------------------------------
// WMMA GEMM:  C[M,Nt] = A[M,K] @ W[Nt,K]^T  (+ epilogue)
//   EPI 0: out_bf16 = relu(C + bias)
//   EPI 1: out_f32  = C + bias + resid
//   EPI 2: out_f32  = relu(C)            (no bias)
// Block = 256 threads (8 waves), 128x128 tile; wave = 32(M) x 64(N).
// A-fragment (16-bit 16x32): lane L: row L&15, K {(L>>4)*8..+7} and {+16..+23}
// B-fragment (32x16): lane L: W-row L&15, K {(L>>4)*8..+7} and {+16..+23}
// C/D: VGPR r -> M = r + 8*(lane>>4), N = lane&15
// ---------------------------------------------------------------------------
template<int EPI>
__global__ __launch_bounds__(256)
void wmma_gemm_kernel(const __bf16* __restrict__ A, const __bf16* __restrict__ W,
                      const float* __restrict__ bias, const float* __restrict__ resid,
                      __bf16* __restrict__ outb, float* __restrict__ outf,
                      int M, int Nt, int K)
{
    const int tid   = threadIdx.x;
    const int lane  = tid & 31;
    const int wave  = tid >> 5;
    const int l16   = lane >> 4;   // 0/1
    const int l15   = lane & 15;
    const int waveM = (wave >> 1) * 32;
    const int waveN = (wave & 1) * 64;
    const int blockM = blockIdx.y * 128;
    const int blockN = blockIdx.x * 128;

    const __bf16* aptr[2];
#pragma unroll
    for (int i = 0; i < 2; ++i) {
        int row = blockM + waveM + i * 16 + l15;
        aptr[i] = A + (size_t)row * K + l16 * 8;
    }
    const __bf16* bptr[4];
#pragma unroll
    for (int j = 0; j < 4; ++j) {
        int row = blockN + waveN + j * 16 + l15;
        bptr[j] = W + (size_t)row * K + l16 * 8;
    }

    v8f zero = {};
    v8f acc[2][4];
#pragma unroll
    for (int i = 0; i < 2; ++i)
#pragma unroll
        for (int j = 0; j < 4; ++j) acc[i][j] = zero;

    for (int k0 = 0; k0 < K; k0 += 32) {
        v16bf a[2], b[4];
#pragma unroll
        for (int i = 0; i < 2; ++i) {
            union { v16bf v; float4 f[2]; } u;
            u.f[0] = *reinterpret_cast<const float4*>(aptr[i] + k0);
            u.f[1] = *reinterpret_cast<const float4*>(aptr[i] + k0 + 16);
            a[i] = u.v;
        }
#pragma unroll
        for (int j = 0; j < 4; ++j) {
            union { v16bf v; float4 f[2]; } u;
            u.f[0] = *reinterpret_cast<const float4*>(bptr[j] + k0);
            u.f[1] = *reinterpret_cast<const float4*>(bptr[j] + k0 + 16);
            b[j] = u.v;
        }
#pragma unroll
        for (int i = 0; i < 2; ++i)
#pragma unroll
            for (int j = 0; j < 4; ++j)
                acc[i][j] = __builtin_amdgcn_wmma_f32_16x16x32_bf16(
                    false, a[i], false, b[j], (short)0, acc[i][j], false, false);
    }

    const int ld = Nt;
#pragma unroll
    for (int i = 0; i < 2; ++i) {
#pragma unroll
        for (int j = 0; j < 4; ++j) {
            int gm0 = blockM + waveM + i * 16 + l16 * 8;
            int gn  = blockN + waveN + j * 16 + l15;
            float bv = (EPI == 2) ? 0.0f : bias[gn];
#pragma unroll
            for (int r = 0; r < 8; ++r) {
                int gm = gm0 + r;
                size_t off = (size_t)gm * ld + gn;
                float v = acc[i][j][r];
                if (EPI == 0) {
                    v = fmaxf(v + bv, 0.0f);
                    outb[off] = (__bf16)v;
                } else if (EPI == 1) {
                    outf[off] = v + bv + resid[off];
                } else {
                    outf[off] = fmaxf(v, 0.0f);
                }
            }
        }
    }
}

// ---------------------------------------------------------------------------
// Row L2-normalize feat (f32) -> fn (bf16). One block per row.
// ---------------------------------------------------------------------------
__global__ __launch_bounds__(256)
void k_rownorm(const float* __restrict__ feat, __bf16* __restrict__ fn, int D)
{
    int row = blockIdx.x;
    int tid = threadIdx.x;
    const float* f = feat + (size_t)row * D;
    float s = 0.0f;
    for (int i = tid; i < D; i += 256) { float v = f[i]; s += v * v; }
    __shared__ float red[256];
    red[tid] = s;
    __syncthreads();
    for (int off = 128; off > 0; off >>= 1) {
        if (tid < off) red[tid] += red[tid + off];
        __syncthreads();
    }
    float inv = 1.0f / (sqrtf(red[0]) + 1e-8f);
    for (int i = tid; i < D; i += 256)
        fn[(size_t)row * D + i] = (__bf16)(f[i] * inv);
}

// ---------------------------------------------------------------------------
// Exact K-th largest per row via bit-pattern binary search (values >= 0,
// so uint order == float order). Row cached in LDS (48KB).
// ---------------------------------------------------------------------------
__global__ __launch_bounds__(256)
void k_topk(const float* __restrict__ sim, float* __restrict__ thr, int n, int k)
{
    __shared__ uint32_t vals[N_];
    __shared__ int cnt;
    int row = blockIdx.x;
    int tid = threadIdx.x;
    const uint32_t* s = reinterpret_cast<const uint32_t*>(sim + (size_t)row * n);
    for (int i = tid; i < n; i += 256) vals[i] = s[i];
    __syncthreads();

    uint32_t lo = 0u, hi = 0x7F800000u;   // [0, +inf)
    for (int it = 0; it < 31; ++it) {
        uint32_t mid = (lo + hi) >> 1;
        if (tid == 0) cnt = 0;
        __syncthreads();
        int c = 0;
        for (int i = tid; i < n; i += 256) c += (vals[i] >= mid) ? 1 : 0;
        atomicAdd(&cnt, c);
        __syncthreads();
        if (cnt >= k) lo = mid; else hi = mid;
        __syncthreads();
    }
    if (tid == 0) thr[row] = __uint_as_float(lo);
}

// ---------------------------------------------------------------------------
// In-place mask + symmetrize: adj = 0.5*(a + a^T), a_ij = sim_ij kept iff
// sim_ij >= thr_i and sim_ij > 0.  64x64 tile pairs via LDS (padded).
// ---------------------------------------------------------------------------
__global__ __launch_bounds__(256)
void k_sym(float* __restrict__ sim, const float* __restrict__ thr, int n)
{
    __shared__ float tA[64][65];
    __shared__ float tB[64][65];
    __shared__ float ti[64], tj[64];

    int T = n >> 6;
    int rem = blockIdx.x, bi = 0;
    while (rem >= T - bi) { rem -= (T - bi); ++bi; }
    int bj = bi + rem;

    int tid = threadIdx.x;
    for (int l = 0; l < 16; ++l) {
        int idx = tid + l * 256;
        int r = idx >> 6, c = idx & 63;
        tA[r][c] = sim[(size_t)(bi * 64 + r) * n + bj * 64 + c];
        tB[r][c] = sim[(size_t)(bj * 64 + r) * n + bi * 64 + c];
    }
    if (tid < 64) { ti[tid] = thr[bi * 64 + tid]; tj[tid] = thr[bj * 64 + tid]; }
    __syncthreads();

    for (int l = 0; l < 16; ++l) {
        int idx = tid + l * 256;
        int r = idx >> 6, c = idx & 63;
        // tile (bi,bj), element (r,c)
        float s1 = tA[r][c];
        float a1 = (s1 > 0.0f && s1 >= ti[r]) ? s1 : 0.0f;
        float s2 = tB[c][r];
        float a2 = (s2 > 0.0f && s2 >= tj[c]) ? s2 : 0.0f;
        sim[(size_t)(bi * 64 + r) * n + bj * 64 + c] = 0.5f * (a1 + a2);
        // tile (bj,bi), element (r,c)
        float s3 = tB[r][c];
        float a3 = (s3 > 0.0f && s3 >= tj[r]) ? s3 : 0.0f;
        float s4 = tA[c][r];
        float a4 = (s4 > 0.0f && s4 >= ti[c]) ? s4 : 0.0f;
        sim[(size_t)(bj * 64 + r) * n + bi * 64 + c] = 0.5f * (a3 + a4);
    }
}

// ---------------------------------------------------------------------------
extern "C" void kernel_launch(void* const* d_in, const int* in_sizes, int n_in,
                              void* d_out, int out_size, void* d_ws, size_t ws_size,
                              hipStream_t stream)
{
    const float* X  = (const float*)d_in[0];
    const float* W1 = (const float*)d_in[1];
    const float* b1 = (const float*)d_in[2];
    const float* W2 = (const float*)d_in[3];
    const float* b2 = (const float*)d_in[4];
    const float* W3 = (const float*)d_in[5];
    const float* b3 = (const float*)d_in[6];
    float* sim = (float*)d_out;

    char* p = (char*)d_ws;
    auto alloc = [&](size_t bytes) -> char* {
        char* r = p; p += (bytes + 255) & ~(size_t)255; return r;
    };
    __bf16* Xb   = (__bf16*)alloc((size_t)N_ * D_ * 2);
    __bf16* W1b  = (__bf16*)alloc((size_t)H_ * D_ * 2);
    __bf16* W2b  = (__bf16*)alloc((size_t)H_ * H_ * 2);
    __bf16* W3b  = (__bf16*)alloc((size_t)D_ * H_ * 2);
    __bf16* h1   = (__bf16*)alloc((size_t)N_ * H_ * 2);
    __bf16* h2   = (__bf16*)alloc((size_t)N_ * H_ * 2);
    float*  feat = (float*) alloc((size_t)N_ * D_ * 4);
    __bf16* fnb  = (__bf16*)alloc((size_t)N_ * D_ * 2);
    float*  thr  = (float*) alloc((size_t)N_ * 4);

    k_cvt<<<4096, 256, 0, stream>>>(X,  Xb,  (size_t)N_ * D_);
    k_cvt<<<2048, 256, 0, stream>>>(W1, W1b, (size_t)H_ * D_);
    k_cvt<<<4096, 256, 0, stream>>>(W2, W2b, (size_t)H_ * H_);
    k_cvt<<<2048, 256, 0, stream>>>(W3, W3b, (size_t)D_ * H_);

    // h1 = relu(X @ W1^T + b1)
    wmma_gemm_kernel<0><<<dim3(H_ / 128, N_ / 128), 256, 0, stream>>>(
        Xb, W1b, b1, nullptr, h1, nullptr, N_, H_, D_);
    // h2 = relu(h1 @ W2^T + b2)
    wmma_gemm_kernel<0><<<dim3(H_ / 128, N_ / 128), 256, 0, stream>>>(
        h1, W2b, b2, nullptr, h2, nullptr, N_, H_, H_);
    // feat = X + (h2 @ W3^T + b3)
    wmma_gemm_kernel<1><<<dim3(D_ / 128, N_ / 128), 256, 0, stream>>>(
        h2, W3b, b3, X, nullptr, feat, N_, D_, H_);
    // fn = feat / (||feat|| + eps)
    k_rownorm<<<N_, 256, 0, stream>>>(feat, fnb, D_);
    // sim = relu(fn @ fn^T)   (written to d_out)
    wmma_gemm_kernel<2><<<dim3(N_ / 128, N_ / 128), 256, 0, stream>>>(
        fnb, fnb, nullptr, nullptr, nullptr, sim, N_, N_, D_);
    // per-row 32nd-largest threshold
    k_topk<<<N_, 256, 0, stream>>>(sim, thr, N_, KTOP);
    // mask + symmetrize in place
    int T = N_ / 64;
    k_sym<<<T * (T + 1) / 2, 256, 0, stream>>>(sim, thr, N_);
}
